// ISDT_19095424598404
// MI455X (gfx1250) — matrix operand order repaired and checked by reading, not compiled, not run on hardware
//
#include <hip/hip_runtime.h>
#include <hip/hip_bf16.h>

#define DEVINL __device__ __forceinline__

typedef __attribute__((ext_vector_type(16))) _Float16 v16h;
typedef __attribute__((ext_vector_type(8)))  _Float16 v8h;
typedef __attribute__((ext_vector_type(8)))  float    v8f;
typedef __attribute__((ext_vector_type(4)))  unsigned int v4u;
typedef __attribute__((ext_vector_type(8)))  unsigned int v8u;

static constexpr int N_ROWS = 131072;
static constexpr int IN_DIM = 512;
static constexpr int HID    = 128;
static constexpr int KCB    = 256;
static constexpr int TOP_M  = 32;

// ---- packed f16 region layout inside d_ws (offsets in halfs) ----
static constexpr int OFF_PW1 = 0;        // 512*128 = 65536 halfs
static constexpr int OFF_PW2 = 65536;    // 128*128 = 16384
static constexpr int OFF_PWM = 81920;
static constexpr int OFF_PWT = 98304;
static constexpr int OFF_PWP = 114688;
static constexpr int OFF_PCM = 131072;   // 128*256 = 32768 each
static constexpr int OFF_PCT = 163840;
static constexpr int OFF_PCP = 196608;
static constexpr int PACK_TOTAL = 229376;               // halfs
static constexpr size_t WSB_INVNORM = (size_t)PACK_TOTAL * 2;   // 458752 B, 768 floats
static constexpr size_t WSB_ALPHA   = WSB_INVNORM + 768 * 4;    // 461824 B, N floats

// ---------------- helpers ----------------
DEVINL v8f splat8(float v) {
  v8f r;
#pragma unroll
  for (int i = 0; i < 8; ++i) r[i] = v;
  return r;
}

DEVINL v16h cat16(v4u lo, v4u hi) {
  v8u r;
  r[0] = lo[0]; r[1] = lo[1]; r[2] = lo[2]; r[3] = lo[3];
  r[4] = hi[0]; r[5] = hi[1]; r[6] = hi[2]; r[7] = hi[3];
  return __builtin_bit_cast(v16h, r);
}

// B fragment (32x16 f16): lanes 0-15 -> col n=lane, K {0..7,16..23}; lanes 16-31 -> K {8..15,24..31}.
// Packed so each lane's 16 halfs are contiguous: two global_load_b128.
DEVINL v16h load_b(const _Float16* __restrict__ pm, int tile, int lane) {
  const _Float16* p = pm + ((size_t)(tile * 32 + lane) << 4);
  v4u lo = *(const v4u*)p;
  v4u hi = *(const v4u*)(p + 8);
  return cat16(lo, hi);
}

// A fragment (16x32 f16) from row-major [16][HID] f16 LDS buffer.
// lane<16: M=lane, K {k0..k0+7, k0+16..k0+23}; lane>=16: M=lane-16, K shifted by 8.
DEVINL v16h load_a_lds(const _Float16* buf, int lane, int kt) {
  const int row = lane & 15;
  const int s = (lane >> 4) * 8;
  const _Float16* p = buf + row * HID + kt * 32 + s;
  v4u lo = *(const v4u*)p;         // halfs k..k+7   (ds_load_b128)
  v4u hi = *(const v4u*)(p + 16);  // halfs k+16..k+23
  return cat16(lo, hi);
}

// A fragment straight from f32 global h0 row (convert to f16 on the fly).
DEVINL v16h load_a_f32(const float* __restrict__ rowp, int kbase) {
  const float4 f0 = *(const float4*)(rowp + kbase);
  const float4 f1 = *(const float4*)(rowp + kbase + 4);
  const float4 f2 = *(const float4*)(rowp + kbase + 16);
  const float4 f3 = *(const float4*)(rowp + kbase + 20);
  v16h a;
  a[0]  = (_Float16)f0.x; a[1]  = (_Float16)f0.y; a[2]  = (_Float16)f0.z; a[3]  = (_Float16)f0.w;
  a[4]  = (_Float16)f1.x; a[5]  = (_Float16)f1.y; a[6]  = (_Float16)f1.z; a[7]  = (_Float16)f1.w;
  a[8]  = (_Float16)f2.x; a[9]  = (_Float16)f2.y; a[10] = (_Float16)f2.z; a[11] = (_Float16)f2.w;
  a[12] = (_Float16)f3.x; a[13] = (_Float16)f3.y; a[14] = (_Float16)f3.z; a[15] = (_Float16)f3.w;
  return a;
}

DEVINL v8f wmma16(v16h a, v16h b, v8f c) {
  // D = A(16x32 f16) * B(32x16 f16) + C(16x16 f32)
  return __builtin_amdgcn_wmma_f32_16x16x32_f16(false, a, false, b, (short)0, c,
                                                false, false);
}

// scatter C-layout accumulator (16x16 f32) into row-major [16][HID] f16 LDS buffer
DEVINL void store_c(_Float16* buf, int lane, int nt, v8f c, bool do_relu) {
  const int col = nt * 16 + (lane & 15);
  const int mb = (lane >> 4) * 8;
#pragma unroll
  for (int r = 0; r < 8; ++r) {
    float v = c[r];
    if (do_relu) v = fmaxf(v, 0.0f);
    buf[(mb + r) * HID + col] = (_Float16)v;
  }
}

// 128x128 GEMM for two 16-row M-tiles: dst = act(src @ W + bias)
// src/dst are per-wave LDS buffers [32][HID]; B pipelined depth-2.
DEVINL void gemm_h128(const _Float16* src, _Float16* dst,
                      const _Float16* __restrict__ pw,
                      const float* __restrict__ bias, bool do_relu, int lane) {
  const int l15 = lane & 15;
  v8f acc[2][8];
#pragma unroll
  for (int nt = 0; nt < 8; ++nt) {
    const float bv = bias[nt * 16 + l15];
    acc[0][nt] = splat8(bv);
    acc[1][nt] = splat8(bv);
  }
  v16h bA = load_b(pw, 0, lane);
  v16h bB = load_b(pw, 1, lane);
  v16h a0 = load_a_lds(src, lane, 0);
  v16h a1 = load_a_lds(src + 16 * HID, lane, 0);
  v16h a0n = a0, a1n = a1;
#pragma unroll
  for (int t = 0; t < 32; ++t) {   // t = kt*8 + nt
    const int kt = t >> 3, nt = t & 7;
    v16h bu = bA;
    bA = bB;
    if (t + 2 < 32) bB = load_b(pw, t + 2, lane);
    if (nt == 0 && kt < 3) {       // prefetch next k-tile's A while computing
      a0n = load_a_lds(src, lane, kt + 1);
      a1n = load_a_lds(src + 16 * HID, lane, kt + 1);
    }
    acc[0][nt] = wmma16(a0, bu, acc[0][nt]);
    acc[1][nt] = wmma16(a1, bu, acc[1][nt]);
    if (nt == 7) { a0 = a0n; a1 = a1n; }
  }
#pragma unroll
  for (int nt = 0; nt < 8; ++nt) {
    store_c(dst, lane, nt, acc[0][nt], do_relu);
    store_c(dst + 16 * HID, lane, nt, acc[1][nt], do_relu);
  }
}

// ---------------- prep kernel 1: codebook row inverse norms ----------------
__global__ __launch_bounds__(128) void cdna5_cb_norms(
    const float* __restrict__ cbm, const float* __restrict__ cbt,
    const float* __restrict__ cbp, float* __restrict__ invnorm) {
  const int b = blockIdx.x;                       // 0..767
  const float* cb = (b < 256) ? cbm : ((b < 512) ? cbt : cbp);
  const int row = b & 255;
  float v = cb[row * HID + threadIdx.x];
  float s = v * v;
#pragma unroll
  for (int m = 16; m > 0; m >>= 1) s += __shfl_xor(s, m, 32);
  __shared__ float ps[4];
  if ((threadIdx.x & 31) == 0) ps[threadIdx.x >> 5] = s;
  __syncthreads();
  if (threadIdx.x == 0) {
    float t = ps[0] + ps[1] + ps[2] + ps[3];
    float nrm = fmaxf(sqrtf(t), 1e-12f);
    invnorm[b] = 1.0f / nrm;
  }
}

// ---------------- prep kernel 2: pack weights & normalized codebooks ----------------
__global__ __launch_bounds__(256) void cdna5_pack_all(
    const float* __restrict__ W1, const float* __restrict__ W2,
    const float* __restrict__ Wm, const float* __restrict__ Wt,
    const float* __restrict__ Wp, const float* __restrict__ cbm,
    const float* __restrict__ cbt, const float* __restrict__ cbp,
    const float* __restrict__ invnorm, _Float16* __restrict__ pack) {
  const int gid = blockIdx.x * 256 + threadIdx.x;   // exactly PACK_TOTAL threads
  if (gid < 65536) {
    // W1: [512,128], ktiles=16, ntiles=8
    const int o = gid;
    const int j = o & 15, lane = (o >> 4) & 31, tile = o >> 9;
    const int nt = tile & 7, kt = tile >> 3;
    const int k = kt * 32 + ((j >> 3) << 4) + ((lane >> 4) << 3) + (j & 7);
    const int n = nt * 16 + (lane & 15);
    pack[gid] = (_Float16)W1[k * HID + n];
  } else if (gid < 131072) {
    // W2/Wm/Wt/Wp: [128,128], ktiles=4, ntiles=8
    const int r = gid - 65536;
    const int mi = r >> 14;
    const int o = r & 16383;
    const float* W = (mi == 0) ? W2 : ((mi == 1) ? Wm : ((mi == 2) ? Wt : Wp));
    const int j = o & 15, lane = (o >> 4) & 31, tile = o >> 9;
    const int nt = tile & 7, kt = tile >> 3;
    const int k = kt * 32 + ((j >> 3) << 4) + ((lane >> 4) << 3) + (j & 7);
    const int n = nt * 16 + (lane & 15);
    pack[gid] = (_Float16)W[k * HID + n];
  } else {
    // normalized codebooks, packed as B = l2n(cb)^T : [128 (k=hid) x 256 (n=code)]
    // ktiles=4, ntiles=16, tile = kt*16 + nt
    const int r = gid - 131072;
    const int ci = r >> 15;
    const int o = r & 32767;
    const float* cb = (ci == 0) ? cbm : ((ci == 1) ? cbt : cbp);
    const int j = o & 15, lane = (o >> 4) & 31, tile = o >> 9;
    const int nt = tile & 15, kt = tile >> 4;
    const int k = kt * 32 + ((j >> 3) << 4) + ((lane >> 4) << 3) + (j & 7);   // hid index
    const int n = nt * 16 + (lane & 15);              // code index
    pack[gid] = (_Float16)(cb[n * HID + k] * invnorm[ci * 256 + n]);
  }
}

// ---------------- main fused kernel ----------------
// 4 waves/block, each wave owns 32 rows (two 16-row M-tiles) end-to-end.
// Two WMMAs per B fragment -> halved weight traffic and load-stall frequency.
__global__ __launch_bounds__(128, 1) void cdna5_fused_encoder(
    const float* __restrict__ h0, const float* __restrict__ b1,
    const float* __restrict__ b2, const float* __restrict__ bm,
    const float* __restrict__ bt, const float* __restrict__ bp,
    const float* __restrict__ ws, const float* __restrict__ wb,
    const _Float16* __restrict__ pk, int* __restrict__ codes,
    float* __restrict__ alpha) {
  __shared__ __attribute__((aligned(16))) _Float16 Hbuf[4][32 * HID];
  __shared__ __attribute__((aligned(16))) _Float16 Zbuf[4][32 * HID];

  const int tid = threadIdx.x;
  const int wave = tid >> 5;
  const int lane = tid & 31;
  const int l15 = lane & 15;
  const int hi = lane >> 4;
  const int R = blockIdx.x * 128 + wave * 32;
  _Float16* hb = Hbuf[wave];
  _Float16* zb = Zbuf[wave];

  // ===== stage 1: H1 = relu(h0 @ W1 + b1), K = 512 (16 k-tiles) =====
  {
    v8f acc[2][8];
#pragma unroll
    for (int nt = 0; nt < 8; ++nt) {
      const float bv = b1[nt * 16 + l15];
      acc[0][nt] = splat8(bv);
      acc[1][nt] = splat8(bv);
    }
    const float* rp0 = h0 + (size_t)(R + l15) * IN_DIM;
    const float* rp1 = rp0 + (size_t)16 * IN_DIM;
    v16h bA = load_b(pk + OFF_PW1, 0, lane);
    v16h bB = load_b(pk + OFF_PW1, 1, lane);
    v16h a0 = load_a_f32(rp0, hi * 8);
    v16h a1 = load_a_f32(rp1, hi * 8);
    v16h a0n = a0, a1n = a1;
#pragma unroll
    for (int t = 0; t < 128; ++t) {  // t = kt*8 + nt
      const int kt = t >> 3, nt = t & 7;
      v16h bu = bA;
      bA = bB;
      if (t + 2 < 128) bB = load_b(pk + OFF_PW1, t + 2, lane);
      if (nt == 0 && kt < 15) {      // prefetch next k-tile's A (8 b128 loads)
        a0n = load_a_f32(rp0, (kt + 1) * 32 + hi * 8);
        a1n = load_a_f32(rp1, (kt + 1) * 32 + hi * 8);
      }
      acc[0][nt] = wmma16(a0, bu, acc[0][nt]);
      acc[1][nt] = wmma16(a1, bu, acc[1][nt]);
      if (nt == 7) { a0 = a0n; a1 = a1n; }
    }
#pragma unroll
    for (int nt = 0; nt < 8; ++nt) {
      store_c(hb, lane, nt, acc[0][nt], true);
      store_c(hb + 16 * HID, lane, nt, acc[1][nt], true);
    }
  }

  // ===== stage 2: H = relu(H1 @ W2 + b2) =====
  gemm_h128(hb, hb, pk + OFF_PW2, b2, true, lane);

  // ===== alpha = sigmoid(H . ws + wb): 8 halfs per ds_load_b128 =====
  {
    float s0 = wb[0], s1 = wb[0];
    const _Float16* hrow0 = hb + l15 * HID;
    const _Float16* hrow1 = hb + (16 + l15) * HID;
    for (int j8 = 0; j8 < HID / 8; ++j8) {
      v8h h0v = __builtin_bit_cast(v8h, *(const v4u*)(hrow0 + j8 * 8));
      v8h h1v = __builtin_bit_cast(v8h, *(const v4u*)(hrow1 + j8 * 8));
#pragma unroll
      for (int e = 0; e < 8; ++e) {
        const float w = ws[j8 * 8 + e];
        s0 += (float)h0v[e] * w;
        s1 += (float)h1v[e] * w;
      }
    }
    if (hi == 0) {
      alpha[R + l15] = 1.0f / (1.0f + __expf(-s0));
      alpha[R + 16 + l15] = 1.0f / (1.0f + __expf(-s1));
    }
  }

  // ===== three VQ heads =====
  const float* bq[3] = {bm, bt, bp};
  const int offW[3] = {OFF_PWM, OFF_PWT, OFF_PWP};
  const int offC[3] = {OFF_PCM, OFF_PCT, OFF_PCP};
  for (int q = 0; q < 3; ++q) {
    // z = H @ Wq + bq
    gemm_h128(hb, zb, pk + offW[q], bq[q], false, lane);

    // S = z @ l2n(cb)^T ; argmin over 256 codes.
    // (row-scaling by 1/||z|| preserves argmin ordering, so z is used raw)
    v16h za[2][4];
#pragma unroll
    for (int kt = 0; kt < 4; ++kt) {
      za[0][kt] = load_a_lds(zb, lane, kt);
      za[1][kt] = load_a_lds(zb + 16 * HID, lane, kt);
    }

    float minv[2][8];
    int mini[2][8];
#pragma unroll
    for (int u = 0; u < 2; ++u)
#pragma unroll
      for (int r = 0; r < 8; ++r) { minv[u][r] = __builtin_inff(); mini[u][r] = 0; }

    const _Float16* pc = pk + offC[q];
    v16h cA = load_b(pc, 0 * 16 + 0, lane);   // tile index = kt*16 + nt
    v16h cB = load_b(pc, 1 * 16 + 0, lane);
    v8f sc0 = splat8(0.0f), sc1 = splat8(0.0f);
#pragma unroll
    for (int t = 0; t < 64; ++t) {   // t = nt*4 + kt
      const int nt = t >> 2, kt = t & 3;
      v16h cu = cA;
      cA = cB;
      if (t + 2 < 64) {
        const int t2 = t + 2;
        cB = load_b(pc, (t2 & 3) * 16 + (t2 >> 2), lane);
      }
      if (kt == 0) { sc0 = splat8(0.0f); sc1 = splat8(0.0f); }
      sc0 = wmma16(za[0][kt], cu, sc0);
      sc1 = wmma16(za[1][kt], cu, sc1);
      if (kt == 3) {
        const int idx = nt * 16 + l15;
#pragma unroll
        for (int r = 0; r < 8; ++r) {
          float v0 = sc0[r];
          if (v0 < minv[0][r]) { minv[0][r] = v0; mini[0][r] = idx; }
          float v1 = sc1[r];
          if (v1 < minv[1][r]) { minv[1][r] = v1; mini[1][r] = idx; }
        }
      }
    }
    // butterfly reduce over the 16 lanes sharing each M row (ties -> lower idx)
#pragma unroll
    for (int m = 1; m < 16; m <<= 1) {
#pragma unroll
      for (int u = 0; u < 2; ++u)
#pragma unroll
        for (int r = 0; r < 8; ++r) {
          float ov = __shfl_xor(minv[u][r], m, 32);
          int oi = __shfl_xor(mini[u][r], m, 32);
          if (ov < minv[u][r] || (ov == minv[u][r] && oi < mini[u][r])) {
            minv[u][r] = ov;
            mini[u][r] = oi;
          }
        }
    }
    if (l15 == 0) {
      const int mb = hi * 8;
#pragma unroll
      for (int u = 0; u < 2; ++u)
#pragma unroll
        for (int r = 0; r < 8; ++r)
          codes[(size_t)(R + u * 16 + mb + r) * 3 + q] = mini[u][r];
    }
  }
}

// ---------------- top-32 by iterative selection (single block) ----------------
__global__ __launch_bounds__(256) void cdna5_topk(
    float* __restrict__ alpha, const int* __restrict__ codes,
    int* __restrict__ key_idx, int* __restrict__ sel) {
  __shared__ float sv[256];
  __shared__ int si[256];
  const int tid = threadIdx.x;
  for (int m = 0; m < TOP_M; ++m) {
    float bv = -__builtin_inff();
    int bi = 0x7fffffff;
    for (int i = tid; i < N_ROWS; i += 256) {
      float v = alpha[i];
      if (v > bv || (v == bv && i < bi)) { bv = v; bi = i; }
    }
    sv[tid] = bv;
    si[tid] = bi;
    __syncthreads();
    for (int s = 128; s > 0; s >>= 1) {
      if (tid < s) {
        float ov = sv[tid + s];
        int oi = si[tid + s];
        if (ov > sv[tid] || (ov == sv[tid] && oi < si[tid])) {
          sv[tid] = ov;
          si[tid] = oi;
        }
      }
      __syncthreads();
    }
    if (tid == 0) {
      const int b = si[0];
      key_idx[m] = b;
      sel[m * 3 + 0] = codes[b * 3 + 0];
      sel[m * 3 + 1] = codes[b * 3 + 1];
      sel[m * 3 + 2] = codes[b * 3 + 2];
      alpha[b] = -__builtin_inff();   // exclude for next pass (alpha fully rewritten each launch)
    }
    __syncthreads();
  }
}

extern "C" void kernel_launch(void* const* d_in, const int* in_sizes, int n_in,
                              void* d_out, int out_size, void* d_ws, size_t ws_size,
                              hipStream_t stream) {
  (void)in_sizes; (void)n_in; (void)out_size; (void)ws_size;
  const float* h0  = (const float*)d_in[0];
  const float* W1  = (const float*)d_in[1];
  const float* b1  = (const float*)d_in[2];
  const float* W2  = (const float*)d_in[3];
  const float* b2  = (const float*)d_in[4];
  const float* Wm  = (const float*)d_in[5];
  const float* bm  = (const float*)d_in[6];
  const float* Wt  = (const float*)d_in[7];
  const float* bt  = (const float*)d_in[8];
  const float* Wp  = (const float*)d_in[9];
  const float* bp  = (const float*)d_in[10];
  const float* cbm = (const float*)d_in[11];
  const float* cbt = (const float*)d_in[12];
  const float* cbp = (const float*)d_in[13];
  const float* ws  = (const float*)d_in[14];
  const float* wb  = (const float*)d_in[15];

  char* wsb = (char*)d_ws;
  _Float16* pack = (_Float16*)wsb;
  float* invnorm = (float*)(wsb + WSB_INVNORM);
  float* alpha = (float*)(wsb + WSB_ALPHA);

  int* out = (int*)d_out;
  int* codes = out;                        // [N,3]
  int* key_idx = out + (size_t)N_ROWS * 3; // [32]
  int* sel = key_idx + TOP_M;              // [32,3]

  cdna5_cb_norms<<<768, 128, 0, stream>>>(cbm, cbt, cbp, invnorm);
  cdna5_pack_all<<<PACK_TOTAL / 256, 256, 0, stream>>>(W1, W2, Wm, Wt, Wp, cbm,
                                                       cbt, cbp, invnorm, pack);
  cdna5_fused_encoder<<<N_ROWS / 128, 128, 0, stream>>>(
      h0, b1, b2, bm, bt, bp, ws, wb, pack, codes, alpha);
  cdna5_topk<<<1, 256, 0, stream>>>(alpha, codes, key_idx, sel);
}
